// DIIN_11974368821250
// MI455X (gfx1250) — matrix operand order repaired and checked by reading, not compiled, not run on hardware
//
#include <hip/hip_runtime.h>
#include <hip/hip_bf16.h>

// ---------------- problem constants ----------------
#define Bq   64
#define Lq   64
#define Cc   16
#define EMB  300
#define CHD  64
#define Dd   450          // EMB + 3*50
#define KP   480          // K padded to multiple of 32
#define NP   464          // N padded to multiple of 16
#define Mrows 4096        // B*L per branch

typedef __attribute__((ext_vector_type(16))) _Float16 v16h;
typedef __attribute__((ext_vector_type(8)))  _Float16 v8h;
typedef __attribute__((ext_vector_type(8)))  float    v8f;

// ---- fragment loaders (CDNA5 WMMA f16 16x16x32 layouts, ISA 7.12.2) ----
// A 16x32: lanes 0-15 row=lane, halves = K{0..7,16..23}; lanes 16-31 K{8..15,24..31}
__device__ __forceinline__ v16h load_frag_a(const _Float16* rowptr, int k0, int lane) {
  int lo = (lane < 16) ? 0 : 8;
  int hi = (lane < 16) ? 16 : 24;
  v16h a;
  *(v8h*)&a       = *(const v8h*)(rowptr + k0 + lo);
  *((v8h*)&a + 1) = *(const v8h*)(rowptr + k0 + hi);
  return a;
}
// B 32x16: lanes 0-15 col=lane, halves = K{0..7,16..23}; lanes 16-31 K{8..15,24..31}
__device__ __forceinline__ v16h load_frag_b(const _Float16* colptr, int k0, int lane) {
  int off = (lane < 16) ? 0 : 8;
  v16h b;
  *(v8h*)&b       = *(const v8h*)(colptr + k0 + off);
  *((v8h*)&b + 1) = *(const v8h*)(colptr + k0 + off + 16);
  return b;
}

// ---------------- 1) pack highway weights f32[450,450] -> f16[464,480] zero padded ----------------
__global__ void pack_w_kernel(const float* __restrict__ hw_w, _Float16* __restrict__ wh) {
  int idx = blockIdx.x * 256 + threadIdx.x;
  if (idx >= NP * KP) return;
  int n = idx / KP, k = idx % KP;
  float v = (n < Dd && k < Dd) ? hw_w[(size_t)n * Dd + k] : 0.0f;
  wh[idx] = (_Float16)v;
}

// ---------------- 2) embedding gather + char CNN ----------------
// one block per token (grid = 2*4096), 256 threads
__global__ __launch_bounds__(256)
void embed_conv_kernel(const int* __restrict__ q1, const int* __restrict__ q2,
                       const int* __restrict__ qc1, const int* __restrict__ qc2,
                       const float* __restrict__ word_emb, const float* __restrict__ char_emb,
                       const float* __restrict__ cw3, const float* __restrict__ cb3,
                       const float* __restrict__ cw4, const float* __restrict__ cb4,
                       const float* __restrict__ cw5, const float* __restrict__ cb5,
                       float* __restrict__ x, _Float16* __restrict__ xh) {
  int t   = threadIdx.x;
  int gid = blockIdx.x;
  int br  = gid >> 12;
  int tok = gid & 4095;
  const int* q  = br ? q2  : q1;
  const int* qc = (br ? qc2 : qc1) + (size_t)tok * Cc;

  __shared__ float e[Cc][CHD];   // e[pos][channel]
  for (int idx = t; idx < Cc * CHD; idx += 256) {
    int pos = idx >> 6, i = idx & 63;
    e[pos][i] = char_emb[(size_t)qc[pos] * CHD + i];
  }
  __syncthreads();

  size_t row = (size_t)br * Mrows + tok;
  float*    xr = x  + row * Dd;
  _Float16* hr = xh + row * KP;

  const float* wrow = word_emb + (size_t)q[tok] * EMB;
  for (int d = t; d < EMB; d += 256) { float v = wrow[d]; xr[d] = v; hr[d] = (_Float16)v; }
  for (int d = Dd + t; d < KP; d += 256) hr[d] = (_Float16)0.0f;

  if (t < 150) {
    const float* W; float bias; int w, f;
    if (t < 50)       { f = t;       W = cw3; bias = cb3[f]; w = 3; }
    else if (t < 100) { f = t - 50;  W = cw4; bias = cb4[f]; w = 4; }
    else              { f = t - 100; W = cw5; bias = cb5[f]; w = 5; }
    const float* Wf = W + (size_t)f * CHD * w;
    int npos = Cc - w + 1;
    float best = -1e30f;
    for (int p = 0; p < npos; ++p) {
      float acc = bias;
      for (int i = 0; i < CHD; ++i) {
        const float* Wi = Wf + i * w;
        for (int ww = 0; ww < w; ++ww) acc += e[p + ww][i] * Wi[ww];
      }
      best = fmaxf(best, acc);
    }
    float out = fmaxf(best, 0.0f);   // max(relu(h)) == relu(max(h))
    xr[EMB + t] = out;
    hr[EMB + t] = (_Float16)out;
  }
}

// ---------------- 3) highway: WMMA GEMM y = x@W^T + b, fused gate ----------------
// grid (256 mtiles, 29 ntiles, 2 branches), 1 wave per 16x16 tile
__global__ __launch_bounds__(32)
void highway_kernel(const _Float16* __restrict__ xh, const float* __restrict__ xin,
                    const _Float16* __restrict__ wh, const float* __restrict__ hw_b,
                    float* __restrict__ xout, _Float16* __restrict__ xho) {
  int lane  = threadIdx.x;
  int mtile = blockIdx.x, ntile = blockIdx.y, br = blockIdx.z;
  int rowg  = br * Mrows + mtile * 16 + (lane & 15);
  int col   = ntile * 16 + (lane & 15);
  const _Float16* ap = xh + (size_t)rowg * KP;
  const _Float16* bp = wh + (size_t)col  * KP;

  v8f c = {};
  for (int k0 = 0; k0 < KP; k0 += 32) {
    v16h a = load_frag_a(ap, k0, lane);
    v16h b = load_frag_b(bp, k0, lane);
    c = __builtin_amdgcn_wmma_f32_16x16x32_f16(false, a, false, b, (short)0, c, false, false);
  }

  int n     = ntile * 16 + (lane & 15);
  int mbase = br * Mrows + mtile * 16 + ((lane < 16) ? 0 : 8);
  float bias = (n < Dd) ? hw_b[n] : 0.0f;
  #pragma unroll
  for (int r = 0; r < 8; ++r) {
    int m = mbase + r;
    if (n < Dd) {
      float y  = c[r] + bias;
      float g  = 1.0f / (1.0f + __expf(-y));
      float rl = fmaxf(y, 0.0f);
      float xo = xin[(size_t)m * Dd + n];
      float o  = g * rl + (1.0f - g) * xo;
      xout[(size_t)m * Dd + n] = o;
      xho[(size_t)m * KP + n]  = (_Float16)o;
    } else {
      xho[(size_t)m * KP + n]  = (_Float16)0.0f;     // cols 450..463
    }
    if (ntile == 28)
      xho[(size_t)m * KP + NP + (lane & 15)] = (_Float16)0.0f;  // cols 464..479
  }
}

// ---------------- 4) attention prep: s1 = x@w1, s2 = x@w2, xa = x*w3 (f16), refresh f16 x ----------------
__global__ __launch_bounds__(64)
void attn_prep_kernel(const float* __restrict__ x, const float* __restrict__ attn_w, int layer,
                      _Float16* __restrict__ xh, _Float16* __restrict__ xa_h,
                      float* __restrict__ s1, float* __restrict__ s2) {
  int t = threadIdx.x;
  size_t row = (size_t)blockIdx.y * Mrows + blockIdx.x;
  const float* w  = attn_w + (size_t)layer * 3 * Dd;
  const float* w1 = w, *w2 = w + Dd, *w3 = w + 2 * Dd;
  const float* xr = x + row * Dd;
  _Float16* hr = xh   + row * KP;
  _Float16* ar = xa_h + row * KP;

  float p1 = 0.0f, p2 = 0.0f;
  for (int d = t; d < KP; d += 64) {
    if (d < Dd) {
      float v = xr[d];
      p1 += v * w1[d];
      p2 += v * w2[d];
      hr[d] = (_Float16)v;
      ar[d] = (_Float16)(v * w3[d]);
    } else {
      hr[d] = (_Float16)0.0f;
      ar[d] = (_Float16)0.0f;
    }
  }
  __shared__ float r1[64], r2[64];
  r1[t] = p1; r2[t] = p2;
  __syncthreads();
  for (int s = 32; s > 0; s >>= 1) {
    if (t < s) { r1[t] += r1[t + s]; r2[t] += r2[t + s]; }
    __syncthreads();
  }
  if (t == 0) { s1[row] = r1[0]; s2[row] = r2[0]; }
}

// ---------------- 5) batched score GEMM: scores[b,i,j] = sum_d (x*w3)[b,i,d]*x[b,j,d] ----------------
// grid (4 itiles, 4 jtiles, 2*64 batch), 1 wave per tile
__global__ __launch_bounds__(32)
void scores_kernel(const _Float16* __restrict__ xa_h, const _Float16* __restrict__ xh,
                   float* __restrict__ scores) {
  int lane  = threadIdx.x;
  int itile = blockIdx.x, jtile = blockIdx.y;
  int bz = blockIdx.z;                    // br*64 + b
  size_t rowbase = (size_t)bz * Lq;       // == br*4096 + b*64
  const _Float16* ap = xa_h + (rowbase + itile * 16 + (lane & 15)) * KP;
  const _Float16* bp = xh   + (rowbase + jtile * 16 + (lane & 15)) * KP;

  v8f c = {};
  for (int k0 = 0; k0 < KP; k0 += 32) {
    v16h a  = load_frag_a(ap, k0, lane);
    v16h bf = load_frag_b(bp, k0, lane);
    c = __builtin_amdgcn_wmma_f32_16x16x32_f16(false, a, false, bf, (short)0, c, false, false);
  }
  int j  = jtile * 16 + (lane & 15);
  int ib = itile * 16 + ((lane < 16) ? 0 : 8);
  float* srow = scores + (size_t)bz * Lq * Lq;
  #pragma unroll
  for (int r = 0; r < 8; ++r)
    srow[(size_t)(ib + r) * Lq + j] = c[r];
}

// ---------------- 6) mask + softmax + AV + residual ----------------
// one block per (branch, b, i); 64 threads = j lanes
__global__ __launch_bounds__(64)
void softmax_av_kernel(const float* __restrict__ scores, const float* __restrict__ s1,
                       const float* __restrict__ s2, const float* __restrict__ attn_b, int layer,
                       const int* __restrict__ len1, const int* __restrict__ len2,
                       const float* __restrict__ x, float* __restrict__ xnext,
                       float* __restrict__ att_out) {
  int t   = threadIdx.x;            // j
  int tok = blockIdx.x;
  int br  = blockIdx.y;
  int b   = tok >> 6, i = tok & 63;
  size_t rowbase = (size_t)br * Mrows + (size_t)b * Lq;
  size_t rowi    = rowbase + i;
  const int* lenp = br ? len2 : len1;
  int   len  = lenp[b];
  float bias = attn_b[layer];

  size_t sbase = ((size_t)(br * Bq + b)) * Lq * Lq + (size_t)i * Lq;
  float sc    = scores[sbase + t] + s1[rowi] + s2[rowbase + t] + bias;
  float logit = (t < len) ? sc : -1e-9f;    // replicate reference's -1e-09 mask value

  __shared__ float red[64];
  __shared__ float p[64];
  red[t] = logit; __syncthreads();
  for (int s = 32; s > 0; s >>= 1) { if (t < s) red[t] = fmaxf(red[t], red[t + s]); __syncthreads(); }
  float mx = red[0]; __syncthreads();
  float e = __expf(logit - mx);
  red[t] = e; __syncthreads();
  for (int s = 32; s > 0; s >>= 1) { if (t < s) red[t] += red[t + s]; __syncthreads(); }
  float denom = red[0];
  p[t] = e / denom; __syncthreads();

  for (int d = t; d < Dd; d += 64) {
    float acc = 0.0f;
    const float* xc = x + rowbase * Dd + d;
    #pragma unroll 8
    for (int j = 0; j < Lq; ++j) acc += p[j] * xc[(size_t)j * Dd];
    att_out[rowi * Dd + d] = acc;
    xnext[rowi * Dd + d]   = acc + x[rowi * Dd + d];
  }
}

// ---------------- host launcher ----------------
extern "C" void kernel_launch(void* const* d_in, const int* in_sizes, int n_in,
                              void* d_out, int out_size, void* d_ws, size_t ws_size,
                              hipStream_t stream) {
  (void)in_sizes; (void)n_in; (void)out_size; (void)ws_size;
  const int*   q1       = (const int*)d_in[0];
  const int*   q2       = (const int*)d_in[1];
  const int*   q1_len   = (const int*)d_in[2];
  const int*   q2_len   = (const int*)d_in[3];
  const int*   q1_char  = (const int*)d_in[4];
  const int*   q2_char  = (const int*)d_in[5];
  const float* word_emb = (const float*)d_in[6];
  const float* char_emb = (const float*)d_in[7];
  const float* cw3 = (const float*)d_in[8],  *cb3 = (const float*)d_in[9];
  const float* cw4 = (const float*)d_in[10], *cb4 = (const float*)d_in[11];
  const float* cw5 = (const float*)d_in[12], *cb5 = (const float*)d_in[13];
  const float* hw_w = (const float*)d_in[14], *hw_b = (const float*)d_in[15];
  const float* attn_w = (const float*)d_in[16], *attn_b = (const float*)d_in[17];
  float* out = (float*)d_out;

  // workspace carve-up (256B aligned)
  char* wp = (char*)d_ws;
  auto carve = [&](size_t bytes) { char* p = wp; wp += (bytes + 255) & ~(size_t)255; return p; };
  _Float16* wh   = (_Float16*)carve((size_t)NP * KP * sizeof(_Float16));
  float*    xA   = (float*)   carve((size_t)2 * Mrows * Dd * sizeof(float));
  float*    xB   = (float*)   carve((size_t)2 * Mrows * Dd * sizeof(float));
  _Float16* hA   = (_Float16*)carve((size_t)2 * Mrows * KP * sizeof(_Float16));
  _Float16* hB   = (_Float16*)carve((size_t)2 * Mrows * KP * sizeof(_Float16));
  _Float16* xa_h = (_Float16*)carve((size_t)2 * Mrows * KP * sizeof(_Float16));
  float*    s1   = (float*)   carve((size_t)2 * Mrows * sizeof(float));
  float*    s2   = (float*)   carve((size_t)2 * Mrows * sizeof(float));
  float*    scb  = (float*)   carve((size_t)2 * Bq * Lq * Lq * sizeof(float));

  // 1) pack highway weights to padded f16
  pack_w_kernel<<<(NP * KP + 255) / 256, 256, 0, stream>>>(hw_w, wh);

  // 2) embedding + char CNN for both branches -> xA (f32), hA (f16 padded)
  embed_conv_kernel<<<2 * Mrows, 256, 0, stream>>>(q1, q2, q1_char, q2_char,
      word_emb, char_emb, cw3, cb3, cw4, cb4, cw5, cb5, xA, hA);

  // 3) two highway layers (WMMA), ping-pong
  dim3 hwGrid(Mrows / 16, (NP / 16), 2);
  highway_kernel<<<hwGrid, 32, 0, stream>>>(hA, xA, wh, hw_b, xB, hB);
  highway_kernel<<<hwGrid, 32, 0, stream>>>(hB, xB, wh, hw_b, xA, hA);

  // 4) attention layer 0: x = xA
  attn_prep_kernel<<<dim3(Mrows, 2), 64, 0, stream>>>(xA, attn_w, 0, hA, xa_h, s1, s2);
  scores_kernel<<<dim3(Lq / 16, Lq / 16, 2 * Bq), 32, 0, stream>>>(xa_h, hA, scb);
  softmax_av_kernel<<<dim3(Mrows, 2), 64, 0, stream>>>(scb, s1, s2, attn_b, 0,
      q1_len, q2_len, xA, xB, out);

  // 5) attention layer 1: x = xB; final att overwrites out (this is the returned tensor pair)
  attn_prep_kernel<<<dim3(Mrows, 2), 64, 0, stream>>>(xB, attn_w, 1, hB, xa_h, s1, s2);
  scores_kernel<<<dim3(Lq / 16, Lq / 16, 2 * Bq), 32, 0, stream>>>(xa_h, hB, scb);
  softmax_av_kernel<<<dim3(Mrows, 2), 64, 0, stream>>>(scb, s1, s2, attn_b, 1,
      q1_len, q2_len, xB, xA, out);
}
